// EntropyCellularPINN_72782515798422
// MI455X (gfx1250) — compile-verified, matrix operand order checked by reading, not run on hardware
//
#include <hip/hip_runtime.h>
#include <math.h>

// ---------------- problem constants ----------------
#define S_TOK 4096
#define DE 1024
#define DH 1024
#define DM 128
#define DC 128
#define DXC (DE + DC) /* 1152 */
#define DEPTH 4
#define SCALE (1.0f / 32.0f) /* 1/sqrt(1024) */

typedef __bf16 bf16_t;
typedef bf16_t v16bf __attribute__((ext_vector_type(16)));
typedef bf16_t v8bf __attribute__((ext_vector_type(8)));
typedef float v8f __attribute__((ext_vector_type(8)));

// ---------------- WMMA helpers ----------------
// 16x32 bf16 operand fragment layout per CDNA5 ISA 7.12.2: lanes 0-15 -> rows
// 0-15 with K = {0..7,16..23}; lanes 16-31 -> same rows with K = {8..15,24..31}.
// GVS addressing: uniform (SGPR) base advances with the K loop; the per-lane
// (row*ld + k-half) term is a loop-invariant 32-bit VGPR offset; tile offsets
// fold into the 24-bit immediate. No per-iteration VALU address updates.
static __device__ __forceinline__ int lane_off(int ld) {
  const int lane = threadIdx.x & 31;
  return (lane & 15) * ld + ((lane >> 4) << 3);
}

static __device__ __forceinline__ v16bf load_frag(const bf16_t* __restrict__ base, int loff) {
  const bf16_t* p = base + loff;
  const v8bf lo = *(const v8bf*)(p);
  const v8bf hi = *(const v8bf*)(p + 16);
  v16bf f;
#pragma unroll
  for (int j = 0; j < 8; ++j) { f[j] = lo[j]; f[8 + j] = hi[j]; }
  return f;
}

static __device__ __forceinline__ v8f wmma_bf16(v16bf a, v16bf b, v8f c) {
  return __builtin_amdgcn_wmma_f32_16x16x32_bf16(false, a, false, b, (short)0, c, false, false);
}

static __device__ __forceinline__ float sigmoidf_(float x) { return 1.0f / (1.0f + __expf(-x)); }

// ---------------- elementwise kernels ----------------
__global__ void cvt_bf16_kernel(const float* __restrict__ src, bf16_t* __restrict__ dst) {
  const size_t i = (size_t)blockIdx.x * 256 + threadIdx.x; // exact multiple grids only
  dst[i] = (bf16_t)src[i];
}

__global__ void concat_x_kernel(const float* __restrict__ e, const float* __restrict__ c,
                                bf16_t* __restrict__ x) {
  const int i = blockIdx.x * 256 + threadIdx.x; // exact multiple: S_TOK*DXC
  const int row = i / DXC, col = i - row * DXC;
  const float v = (col < DE) ? e[(size_t)row * DE + col] : c[(size_t)row * DC + (col - DE)];
  x[i] = (bf16_t)v;
}

__global__ void e_add_kernel(float* __restrict__ e, const float* __restrict__ ht) {
  const size_t i = (size_t)blockIdx.x * 256 + threadIdx.x; // exact multiple: S_TOK*DE
  e[i] += ht[i];
}

// ---------------- generic all-bf16 WMMA GEMM ----------------
// C[M,N] = A[M,K] @ B[N,K]^T + bias ; one wave per block; 32x64 output tile.
// Ping-pong (X/Y) double-buffered fragments, two K-slices per iteration.
// Requires K % 64 == 0 (all call sites comply).
// flags: 1=sigmoid, 2=write C (f32), 4=write Ct (bf16 transposed), 8=write Cb (bf16 row-major)
__global__ void gemm_bb_kernel(const bf16_t* __restrict__ A, int lda,
                               const bf16_t* __restrict__ B, int ldb,
                               const float* __restrict__ bias,
                               float* __restrict__ C, int ldc,
                               bf16_t* __restrict__ Cb, int ldcb,
                               bf16_t* __restrict__ Ct, int ldct,
                               int K, int flags) {
  const int m0 = blockIdx.y * 32;
  const int n0 = blockIdx.x * 64;
  v8f acc[2][4];
#pragma unroll
  for (int mi = 0; mi < 2; ++mi)
#pragma unroll
    for (int t = 0; t < 4; ++t)
#pragma unroll
      for (int r = 0; r < 8; ++r) acc[mi][t][r] = 0.0f;

  const bf16_t* Abase = A + (size_t)m0 * lda; // uniform
  const bf16_t* Bbase = B + (size_t)n0 * ldb; // uniform
  const int aoff = lane_off(lda);             // invariant per-lane VGPR offset
  const int boff = lane_off(ldb);
  const size_t sA = (size_t)16 * lda;         // second m-tile (immediate)

  v16bf xa0, xa1, xb[4], ya0, ya1, yb[4];
  // prologue: X <- slice 0
  xa0 = load_frag(Abase, aoff);
  xa1 = load_frag(Abase + sA, aoff);
#pragma unroll
  for (int t = 0; t < 4; ++t) xb[t] = load_frag(Bbase + (size_t)(16 * t) * ldb, boff);

  int kk = 32;
  for (; kk + 64 < K; kk += 64) {
    __builtin_prefetch((const void*)(Abase + kk + 64), 0, 2);
    __builtin_prefetch((const void*)(Bbase + kk + 64), 0, 2);
    // Y <- slice @kk (in flight while X computes)
    ya0 = load_frag(Abase + kk, aoff);
    ya1 = load_frag(Abase + sA + kk, aoff);
#pragma unroll
    for (int t = 0; t < 4; ++t) yb[t] = load_frag(Bbase + (size_t)(16 * t) * ldb + kk, boff);
#pragma unroll
    for (int t = 0; t < 4; ++t) acc[0][t] = wmma_bf16(xa0, xb[t], acc[0][t]);
#pragma unroll
    for (int t = 0; t < 4; ++t) acc[1][t] = wmma_bf16(xa1, xb[t], acc[1][t]);
    // X <- slice @kk+32 (in flight while Y computes)
    xa0 = load_frag(Abase + kk + 32, aoff);
    xa1 = load_frag(Abase + sA + kk + 32, aoff);
#pragma unroll
    for (int t = 0; t < 4; ++t) xb[t] = load_frag(Bbase + (size_t)(16 * t) * ldb + kk + 32, boff);
#pragma unroll
    for (int t = 0; t < 4; ++t) acc[0][t] = wmma_bf16(ya0, yb[t], acc[0][t]);
#pragma unroll
    for (int t = 0; t < 4; ++t) acc[1][t] = wmma_bf16(ya1, yb[t], acc[1][t]);
  }
  // epilogue: final slice @K-32 into Y, then drain X and Y
  ya0 = load_frag(Abase + (K - 32), aoff);
  ya1 = load_frag(Abase + sA + (K - 32), aoff);
#pragma unroll
  for (int t = 0; t < 4; ++t) yb[t] = load_frag(Bbase + (size_t)(16 * t) * ldb + (K - 32), boff);
#pragma unroll
  for (int t = 0; t < 4; ++t) acc[0][t] = wmma_bf16(xa0, xb[t], acc[0][t]);
#pragma unroll
  for (int t = 0; t < 4; ++t) acc[1][t] = wmma_bf16(xa1, xb[t], acc[1][t]);
#pragma unroll
  for (int t = 0; t < 4; ++t) acc[0][t] = wmma_bf16(ya0, yb[t], acc[0][t]);
#pragma unroll
  for (int t = 0; t < 4; ++t) acc[1][t] = wmma_bf16(ya1, yb[t], acc[1][t]);

  const int lane = threadIdx.x & 31;
  const int cl = lane & 15;
  const int rh = (lane >> 4) * 8;
#pragma unroll
  for (int mi = 0; mi < 2; ++mi) {
    const int mbase = m0 + 16 * mi;
#pragma unroll
    for (int t = 0; t < 4; ++t) {
      const int col = n0 + 16 * t + cl;
      const float bv = bias ? bias[col] : 0.0f;
      v8bf tv;
#pragma unroll
      for (int r = 0; r < 8; ++r) {
        const int row = mbase + r + rh;
        float v = acc[mi][t][r] + bv;
        if (flags & 1) v = sigmoidf_(v);
        if (flags & 2) C[(size_t)row * ldc + col] = v;
        if (flags & 8) Cb[(size_t)row * ldcb + col] = (bf16_t)v;
        tv[r] = (bf16_t)v;
      }
      if (flags & 4) *(v8bf*)(Ct + (size_t)col * ldct + mbase + rh) = tv; // contiguous 16B store
    }
  }
}

// ---------------- masked scores: P = (q k^T * scale) * (m m^T) * (l x l) ----------------
__global__ void scores_kernel(const bf16_t* __restrict__ Q, const bf16_t* __restrict__ Kf,
                              const bf16_t* __restrict__ M, const float* __restrict__ L,
                              float* __restrict__ P) {
  const int m0 = blockIdx.y * 16;
  const int n0 = blockIdx.x * 64;
  v8f accS[4], accM[4];
#pragma unroll
  for (int t = 0; t < 4; ++t)
#pragma unroll
    for (int r = 0; r < 8; ++r) { accS[t][r] = 0.0f; accM[t][r] = 0.0f; }

  const bf16_t* Qbase = Q + (size_t)m0 * DE;
  const bf16_t* Kbase = Kf + (size_t)n0 * DE;
  const int qoff = lane_off(DE);

  // ping-pong q.k^T over K = DE (DE/32 = 32 slices, even)
  v16bf xa, xb[4], ya, yb[4];
  xa = load_frag(Qbase, qoff);
#pragma unroll
  for (int t = 0; t < 4; ++t) xb[t] = load_frag(Kbase + (size_t)(16 * t) * DE, qoff);
  int kk = 32;
  for (; kk + 64 < DE; kk += 64) {
    __builtin_prefetch((const void*)(Qbase + kk + 64), 0, 2);
    __builtin_prefetch((const void*)(Kbase + kk + 64), 0, 2);
    ya = load_frag(Qbase + kk, qoff);
#pragma unroll
    for (int t = 0; t < 4; ++t) yb[t] = load_frag(Kbase + (size_t)(16 * t) * DE + kk, qoff);
#pragma unroll
    for (int t = 0; t < 4; ++t) accS[t] = wmma_bf16(xa, xb[t], accS[t]);
    xa = load_frag(Qbase + kk + 32, qoff);
#pragma unroll
    for (int t = 0; t < 4; ++t) xb[t] = load_frag(Kbase + (size_t)(16 * t) * DE + kk + 32, qoff);
#pragma unroll
    for (int t = 0; t < 4; ++t) accS[t] = wmma_bf16(ya, yb[t], accS[t]);
  }
  ya = load_frag(Qbase + (DE - 32), qoff);
#pragma unroll
  for (int t = 0; t < 4; ++t) yb[t] = load_frag(Kbase + (size_t)(16 * t) * DE + (DE - 32), qoff);
#pragma unroll
  for (int t = 0; t < 4; ++t) accS[t] = wmma_bf16(xa, xb[t], accS[t]);
#pragma unroll
  for (int t = 0; t < 4; ++t) accS[t] = wmma_bf16(ya, yb[t], accS[t]);

  // membrane mask m.m^T over K = DM (batched loads then wmmas per slice)
  const bf16_t* Mabase = M + (size_t)m0 * DM;
  const bf16_t* Mbbase = M + (size_t)n0 * DM;
  const int moff = lane_off(DM);
#pragma unroll
  for (int km = 0; km < DM; km += 32) {
    v16bf am = load_frag(Mabase + km, moff);
    v16bf bm[4];
#pragma unroll
    for (int t = 0; t < 4; ++t) bm[t] = load_frag(Mbbase + (size_t)(16 * t) * DM + km, moff);
#pragma unroll
    for (int t = 0; t < 4; ++t) accM[t] = wmma_bf16(am, bm[t], accM[t]);
  }

  const int lane = threadIdx.x & 31;
  const int cl = lane & 15;
  const int rh = (lane >> 4) * 8;
  float lr[8];
#pragma unroll
  for (int r = 0; r < 8; ++r) lr[r] = L[m0 + r + rh];
#pragma unroll
  for (int t = 0; t < 4; ++t) {
    const int col = n0 + 16 * t + cl;
    const float lc = L[col];
#pragma unroll
    for (int r = 0; r < 8; ++r) {
      const int row = m0 + r + rh;
      P[(size_t)row * S_TOK + col] = accS[t][r] * SCALE * accM[t][r] * lr[r] * lc;
    }
  }
}

// ---------------- row softmax: Pb = softmax(P, axis=-1) as bf16 ----------------
__global__ void softmax_kernel(const float* __restrict__ P, bf16_t* __restrict__ Pb) {
  __shared__ float sh[8];
  __shared__ float bcast;
  const int row = blockIdx.x;
  const int tid = threadIdx.x; // 256 threads
  const float* pr = P + (size_t)row * S_TOK;
  bf16_t* pw = Pb + (size_t)row * S_TOK;

  float mx = -3.402823466e38f;
  for (int j = tid; j < S_TOK; j += 256) mx = fmaxf(mx, pr[j]);
#pragma unroll
  for (int off = 16; off; off >>= 1) mx = fmaxf(mx, __shfl_xor(mx, off, 32));
  if ((tid & 31) == 0) sh[tid >> 5] = mx;
  __syncthreads();
  if (tid == 0) {
    float v = sh[0];
    for (int w = 1; w < 8; ++w) v = fmaxf(v, sh[w]);
    bcast = v;
  }
  __syncthreads();
  mx = bcast;
  __syncthreads();

  float s = 0.0f;
  for (int j = tid; j < S_TOK; j += 256) s += __expf(pr[j] - mx);
#pragma unroll
  for (int off = 16; off; off >>= 1) s += __shfl_xor(s, off, 32);
  if ((tid & 31) == 0) sh[tid >> 5] = s;
  __syncthreads();
  if (tid == 0) {
    float v = 0.0f;
    for (int w = 0; w < 8; ++w) v += sh[w];
    bcast = 1.0f / v;
  }
  __syncthreads();
  const float inv = bcast;

  for (int j = tid; j < S_TOK; j += 256) pw[j] = (bf16_t)(__expf(pr[j] - mx) * inv);
}

// ---------------- fused GRU cell: h_new = GRU(h_tilde, h_prev) ----------------
// One wave per 16x32 tile of [S, DH]; 12 WMMA accumulator chains.
// Batched loads (uniform bases + invariant lane offsets + immediate row
// offsets) then WMMAs.
__global__ void gru_kernel(const bf16_t* __restrict__ Xt, const bf16_t* __restrict__ Hb,
                           const float* __restrict__ Hp,
                           const bf16_t* __restrict__ Wih, const bf16_t* __restrict__ Whh,
                           const float* __restrict__ bih, const float* __restrict__ bhh,
                           float* __restrict__ Hn) {
  const int m0 = blockIdx.y * 16;
  const int n0 = blockIdx.x * 32;
  v8f gi[3][2], gh[3][2];
#pragma unroll
  for (int g = 0; g < 3; ++g)
#pragma unroll
    for (int t = 0; t < 2; ++t)
#pragma unroll
      for (int r = 0; r < 8; ++r) { gi[g][t][r] = 0.0f; gh[g][t][r] = 0.0f; }

  const bf16_t* Xbase = Xt + (size_t)m0 * DE;
  const bf16_t* Hbase = Hb + (size_t)m0 * DH;
  const bf16_t* Wibase = Wih + (size_t)n0 * DE;
  const bf16_t* Whbase = Whh + (size_t)n0 * DH;
  const int xoff = lane_off(DE); // DE == DH: same offset for all four streams

  for (int kk = 0; kk < DE; kk += 32) {
    if (kk + 32 < DE) {
      __builtin_prefetch((const void*)(Xbase + kk + 32), 0, 2);
      __builtin_prefetch((const void*)(Wibase + kk + 32), 0, 2);
    }
    v16bf ax = load_frag(Xbase + kk, xoff);
    v16bf ah = load_frag(Hbase + kk, xoff);
    v16bf bi[3][2], bh[3][2];
#pragma unroll
    for (int g = 0; g < 3; ++g)
#pragma unroll
      for (int t = 0; t < 2; ++t) {
        bi[g][t] = load_frag(Wibase + (size_t)(g * DH + 16 * t) * DE + kk, xoff);
        bh[g][t] = load_frag(Whbase + (size_t)(g * DH + 16 * t) * DH + kk, xoff);
      }
#pragma unroll
    for (int g = 0; g < 3; ++g)
#pragma unroll
      for (int t = 0; t < 2; ++t) {
        gi[g][t] = wmma_bf16(ax, bi[g][t], gi[g][t]);
        gh[g][t] = wmma_bf16(ah, bh[g][t], gh[g][t]);
      }
  }

  const int lane = threadIdx.x & 31;
  const int cl = lane & 15;
  const int rh = (lane >> 4) * 8;
#pragma unroll
  for (int t = 0; t < 2; ++t) {
    const int col = n0 + 16 * t + cl;
#pragma unroll
    for (int r = 0; r < 8; ++r) {
      const int row = m0 + r + rh;
      const float ir = gi[0][t][r] + bih[col];
      const float iz = gi[1][t][r] + bih[DH + col];
      const float in_ = gi[2][t][r] + bih[2 * DH + col];
      const float hr = gh[0][t][r] + bhh[col];
      const float hz = gh[1][t][r] + bhh[DH + col];
      const float hn = gh[2][t][r] + bhh[2 * DH + col];
      const float rg = sigmoidf_(ir + hr);
      const float zg = sigmoidf_(iz + hz);
      const float ng = tanhf(in_ + rg * hn);
      const float hp = Hp[(size_t)row * DH + col];
      Hn[(size_t)row * DH + col] = (1.0f - zg) * ng + zg * hp;
    }
  }
}

// ---------------- lifespan update + h commit (f32 + bf16 shadow) ----------------
__global__ void hl_update_kernel(const float* __restrict__ Hn, float* __restrict__ Hc,
                                 bf16_t* __restrict__ Hcb, float* __restrict__ l) {
  __shared__ float sh[8];
  const int row = blockIdx.x;
  const int tid = threadIdx.x;
  const float* pn = Hn + (size_t)row * DH;
  float* pc = Hc + (size_t)row * DH;
  bf16_t* pb = Hcb + (size_t)row * DH;
  float s = 0.0f;
  for (int j = tid; j < DH; j += 256) {
    const float d = pn[j] - pc[j];
    s += d * d;
  }
#pragma unroll
  for (int off = 16; off; off >>= 1) s += __shfl_xor(s, off, 32);
  if ((tid & 31) == 0) sh[tid >> 5] = s;
  __syncthreads();
  if (tid == 0) {
    float tot = 0.0f;
    for (int w = 0; w < 8; ++w) tot += sh[w];
    const float hd = sqrtf(tot);
    const float sat = (hd < 1e-4f) ? 1.0f : 0.0f;
    const float lv = l[row] - 0.1f * (1.0f + 5.0f * sat);
    l[row] = fminf(fmaxf(lv, 0.0f), 1.0f);
  }
  __syncthreads();
  for (int j = tid; j < DH; j += 256) {
    const float v = pn[j];
    pc[j] = v;
    pb[j] = (bf16_t)v;
  }
}

__global__ void num_died_kernel(const float* __restrict__ l, float* __restrict__ out) {
  __shared__ float sh[8];
  const int tid = threadIdx.x;
  float cnt = 0.0f;
  for (int j = tid; j < S_TOK; j += 256) cnt += (l[j] <= 0.05f) ? 1.0f : 0.0f;
#pragma unroll
  for (int off = 16; off; off >>= 1) cnt += __shfl_xor(cnt, off, 32);
  if ((tid & 31) == 0) sh[tid >> 5] = cnt;
  __syncthreads();
  if (tid == 0) {
    float s = 0.0f;
    for (int w = 0; w < 8; ++w) s += sh[w];
    *out = s;
  }
}

// ---------------- host orchestration ----------------
extern "C" void kernel_launch(void* const* d_in, const int* in_sizes, int n_in,
                              void* d_out, int out_size, void* d_ws, size_t ws_size,
                              hipStream_t stream) {
  (void)in_sizes; (void)n_in; (void)out_size; (void)ws_size;
  const float* in_e = (const float*)d_in[0];
  const float* in_h = (const float*)d_in[1];
  const float* in_m = (const float*)d_in[2];
  const float* in_c = (const float*)d_in[3];
  const float* in_l = (const float*)d_in[4];
  const float* Wq = (const float*)d_in[5];
  const float* bq = (const float*)d_in[6];
  const float* Wk = (const float*)d_in[7];
  const float* bk = (const float*)d_in[8];
  const float* Wv = (const float*)d_in[9];
  const float* bv = (const float*)d_in[10];
  const float* Wmem = (const float*)d_in[11];
  const float* bmem = (const float*)d_in[12];
  const float* Wih = (const float*)d_in[13];
  const float* Whh = (const float*)d_in[14];
  const float* bih = (const float*)d_in[15];
  const float* bhh = (const float*)d_in[16];
  float* out = (float*)d_out;

  char* wsb = (char*)d_ws;
  size_t off = 0;
  auto alloc = [&](size_t bytes) -> void* {
    void* p = wsb + off;
    off += (bytes + 255) & ~(size_t)255;
    return p;
  };
  // f32 state
  float* e_cur = (float*)alloc((size_t)S_TOK * DE * 4);
  float* h_cur = (float*)alloc((size_t)S_TOK * DH * 4);
  float* m_cur = (float*)alloc((size_t)S_TOK * DM * 4);
  float* l_cur = (float*)alloc((size_t)S_TOK * 4);
  float* h_til = (float*)alloc((size_t)S_TOK * DE * 4);
  float* h_new = (float*)alloc((size_t)S_TOK * DH * 4);
  float* P     = (float*)alloc((size_t)S_TOK * S_TOK * 4);
  // bf16 activations
  bf16_t* xb  = (bf16_t*)alloc((size_t)S_TOK * DXC * 2);
  bf16_t* qb  = (bf16_t*)alloc((size_t)S_TOK * DE * 2);
  bf16_t* kb  = (bf16_t*)alloc((size_t)S_TOK * DE * 2);
  bf16_t* vt  = (bf16_t*)alloc((size_t)DE * S_TOK * 2);
  bf16_t* mb  = (bf16_t*)alloc((size_t)S_TOK * DM * 2);
  bf16_t* hb  = (bf16_t*)alloc((size_t)S_TOK * DH * 2);
  bf16_t* htb = (bf16_t*)alloc((size_t)S_TOK * DE * 2);
  bf16_t* Pb  = (bf16_t*)alloc((size_t)S_TOK * S_TOK * 2);
  // bf16 weights (converted once per call)
  bf16_t* Wq_b  = (bf16_t*)alloc((size_t)DEPTH * DE * DXC * 2);
  bf16_t* Wk_b  = (bf16_t*)alloc((size_t)DEPTH * DE * DXC * 2);
  bf16_t* Wv_b  = (bf16_t*)alloc((size_t)DEPTH * DE * DXC * 2);
  bf16_t* Wm_b  = (bf16_t*)alloc((size_t)DEPTH * DM * DH * 2);
  bf16_t* Wih_b = (bf16_t*)alloc((size_t)DEPTH * 3 * DH * DE * 2);
  bf16_t* Whh_b = (bf16_t*)alloc((size_t)DEPTH * 3 * DH * DH * 2);

  const dim3 blk32(32), blk256(256);

  // one-time conversions (deterministic, every call)
  cvt_bf16_kernel<<<(DEPTH * DE * DXC) / 256, blk256, 0, stream>>>(Wq, Wq_b);
  cvt_bf16_kernel<<<(DEPTH * DE * DXC) / 256, blk256, 0, stream>>>(Wk, Wk_b);
  cvt_bf16_kernel<<<(DEPTH * DE * DXC) / 256, blk256, 0, stream>>>(Wv, Wv_b);
  cvt_bf16_kernel<<<(DEPTH * DM * DH) / 256, blk256, 0, stream>>>(Wmem, Wm_b);
  cvt_bf16_kernel<<<(DEPTH * 3 * DH * DE) / 256, blk256, 0, stream>>>(Wih, Wih_b);
  cvt_bf16_kernel<<<(DEPTH * 3 * DH * DH) / 256, blk256, 0, stream>>>(Whh, Whh_b);
  cvt_bf16_kernel<<<(S_TOK * DH) / 256, blk256, 0, stream>>>(in_h, hb);
  cvt_bf16_kernel<<<(S_TOK * DM) / 256, blk256, 0, stream>>>(in_m, mb);

  hipMemcpyAsync(e_cur, in_e, (size_t)S_TOK * DE * 4, hipMemcpyDeviceToDevice, stream);
  hipMemcpyAsync(h_cur, in_h, (size_t)S_TOK * DH * 4, hipMemcpyDeviceToDevice, stream);
  hipMemcpyAsync(l_cur, in_l, (size_t)S_TOK * 4, hipMemcpyDeviceToDevice, stream);

  const dim3 grid_gemm_de(DE / 64, S_TOK / 32);
  const dim3 grid_gemm_dm(DM / 64, S_TOK / 32);
  const dim3 grid_scores(S_TOK / 64, S_TOK / 16);
  const dim3 grid_gru(DH / 32, S_TOK / 16);

  for (int i = 0; i < DEPTH; ++i) {
    const bf16_t* Wq_i = Wq_b + (size_t)i * DE * DXC;
    const bf16_t* Wk_i = Wk_b + (size_t)i * DE * DXC;
    const bf16_t* Wv_i = Wv_b + (size_t)i * DE * DXC;
    const bf16_t* Wm_i = Wm_b + (size_t)i * DM * DH;
    const bf16_t* Wih_i = Wih_b + (size_t)i * 3 * DH * DE;
    const bf16_t* Whh_i = Whh_b + (size_t)i * 3 * DH * DH;
    const float* bq_i = bq + (size_t)i * DE;
    const float* bk_i = bk + (size_t)i * DE;
    const float* bv_i = bv + (size_t)i * DE;
    const float* bm_i = bmem + (size_t)i * DM;
    const float* bih_i = bih + (size_t)i * 3 * DH;
    const float* bhh_i = bhh + (size_t)i * 3 * DH;

    // x = concat(e, c) -> bf16
    concat_x_kernel<<<(S_TOK * DXC) / 256, blk256, 0, stream>>>(e_cur, in_c, xb);
    // q, k -> bf16 row-major; v -> bf16 transposed
    gemm_bb_kernel<<<grid_gemm_de, blk32, 0, stream>>>(xb, DXC, Wq_i, DXC, bq_i,
                                                       nullptr, 0, qb, DE, nullptr, 0, DXC, 8);
    gemm_bb_kernel<<<grid_gemm_de, blk32, 0, stream>>>(xb, DXC, Wk_i, DXC, bk_i,
                                                       nullptr, 0, kb, DE, nullptr, 0, DXC, 8);
    gemm_bb_kernel<<<grid_gemm_de, blk32, 0, stream>>>(xb, DXC, Wv_i, DXC, bv_i,
                                                       nullptr, 0, nullptr, 0, vt, S_TOK, DXC, 4);
    // masked scores, softmax, h_tilde = P @ V
    scores_kernel<<<grid_scores, blk32, 0, stream>>>(qb, kb, mb, l_cur, P);
    softmax_kernel<<<S_TOK, blk256, 0, stream>>>(P, Pb);
    gemm_bb_kernel<<<grid_gemm_de, blk32, 0, stream>>>(Pb, S_TOK, vt, S_TOK, nullptr,
                                                       h_til, DE, htb, DE, nullptr, 0, S_TOK, 2 | 8);
    // e = e + h_tilde
    e_add_kernel<<<(S_TOK * DE) / 256, blk256, 0, stream>>>(e_cur, h_til);
    // GRU, memory gate, lifespan
    gru_kernel<<<grid_gru, blk32, 0, stream>>>(htb, hb, h_cur, Wih_i, Whh_i, bih_i, bhh_i, h_new);
    cvt_bf16_kernel<<<(S_TOK * DH) / 256, blk256, 0, stream>>>(h_new, htb); // reuse htb as h_new bf16
    gemm_bb_kernel<<<grid_gemm_dm, blk32, 0, stream>>>(htb, DH, Wm_i, DH, bm_i,
                                                       m_cur, DM, mb, DM, nullptr, 0, DH, 1 | 2 | 8);
    hl_update_kernel<<<S_TOK, blk256, 0, stream>>>(h_new, h_cur, hb, l_cur);
  }

  // pack outputs: e, h, m, c, l, num_died
  const size_t OUT_E = 0;
  const size_t OUT_H = OUT_E + (size_t)S_TOK * DE;
  const size_t OUT_M = OUT_H + (size_t)S_TOK * DH;
  const size_t OUT_C = OUT_M + (size_t)S_TOK * DM;
  const size_t OUT_L = OUT_C + (size_t)S_TOK * DC;
  const size_t OUT_ND = OUT_L + (size_t)S_TOK;
  hipMemcpyAsync(out + OUT_E, e_cur, (size_t)S_TOK * DE * 4, hipMemcpyDeviceToDevice, stream);
  hipMemcpyAsync(out + OUT_H, h_cur, (size_t)S_TOK * DH * 4, hipMemcpyDeviceToDevice, stream);
  hipMemcpyAsync(out + OUT_M, m_cur, (size_t)S_TOK * DM * 4, hipMemcpyDeviceToDevice, stream);
  hipMemcpyAsync(out + OUT_C, in_c, (size_t)S_TOK * DC * 4, hipMemcpyDeviceToDevice, stream);
  hipMemcpyAsync(out + OUT_L, l_cur, (size_t)S_TOK * 4, hipMemcpyDeviceToDevice, stream);
  num_died_kernel<<<1, blk256, 0, stream>>>(l_cur, out + OUT_ND);
}
